// MessagePassingLayer_10462540333519
// MI455X (gfx1250) — compile-verified, matrix operand order checked by reading, not run on hardware
//
#include <hip/hip_runtime.h>

#define Bdim 64
#define Kdim 32
#define Ldim 64
#define Hdim 256

typedef __bf16 bf16;
typedef __attribute__((ext_vector_type(16))) __bf16 v16bf;
typedef __attribute__((ext_vector_type(8)))  __bf16 v8bf;
typedef __attribute__((ext_vector_type(8)))  float  v8f;

// ---------------------------------------------------------------------------
// WMMA helpers (V_WMMA_F32_16X16X32_BF16 layouts, wave32)
//   lane = half*16 + idx ; half = lane>>4, idx = lane&15
//   A frag element j: j<8  -> K = half*8 + j
//                     j>=8 -> K = 16 + half*8 + (j-8)
//   (two contiguous 8-element groups per half -> b128 loads)
//   B frag: col n = idx ; same K mapping (reads W[o][k] contiguously)
//   C (v8f): M = v + 8*half, N = idx
// ---------------------------------------------------------------------------
__device__ __forceinline__ v16bf load_a_f32(const float* __restrict__ row,
                                            int kk, int half, float alpha) {
  const float* p0 = row + kk + half * 8;
  const float* p1 = row + kk + 16 + half * 8;
  float4 f0 = *reinterpret_cast<const float4*>(p0);
  float4 f1 = *reinterpret_cast<const float4*>(p0 + 4);
  float4 f2 = *reinterpret_cast<const float4*>(p1);
  float4 f3 = *reinterpret_cast<const float4*>(p1 + 4);
  v16bf a;
  a[0]  = (bf16)(alpha * f0.x); a[1]  = (bf16)(alpha * f0.y);
  a[2]  = (bf16)(alpha * f0.z); a[3]  = (bf16)(alpha * f0.w);
  a[4]  = (bf16)(alpha * f1.x); a[5]  = (bf16)(alpha * f1.y);
  a[6]  = (bf16)(alpha * f1.z); a[7]  = (bf16)(alpha * f1.w);
  a[8]  = (bf16)(alpha * f2.x); a[9]  = (bf16)(alpha * f2.y);
  a[10] = (bf16)(alpha * f2.z); a[11] = (bf16)(alpha * f2.w);
  a[12] = (bf16)(alpha * f3.x); a[13] = (bf16)(alpha * f3.y);
  a[14] = (bf16)(alpha * f3.z); a[15] = (bf16)(alpha * f3.w);
  return a;
}

__device__ __forceinline__ v16bf load_frag_bf16(const bf16* __restrict__ row,
                                                int kk, int half) {
  v8bf u0 = *reinterpret_cast<const v8bf*>(row + kk + half * 8);
  v8bf u1 = *reinterpret_cast<const v8bf*>(row + kk + 16 + half * 8);
  v16bf a;
#pragma unroll
  for (int j = 0; j < 8; ++j) {
    a[j]     = u0[j];
    a[8 + j] = u1[j];
  }
  return a;
}

__device__ __forceinline__ v8f wmma_bf16(v16bf a, v16bf b, v8f c) {
  return __builtin_amdgcn_wmma_f32_16x16x32_bf16(false, a, false, b,
                                                 (short)0, c, false, false);
}

// ---------------------------------------------------------------------------
// Utility kernels
// ---------------------------------------------------------------------------
__global__ void zero_kernel(float* __restrict__ p, int n) {
  int i = blockIdx.x * blockDim.x + threadIdx.x;
  if (i < n) p[i] = 0.0f;
}

__global__ void cvt_bf16_kernel(const float* __restrict__ src,
                                bf16* __restrict__ dst, int n) {
  int i = blockIdx.x * blockDim.x + threadIdx.x;
  if (i < n) dst[i] = (bf16)src[i];
}

// ---------------------------------------------------------------------------
// Generic GEMM: out[r][o] = alpha * (A[r][:] . W[o][coloff:]) + bias[o]
// One wave per 16x64 tile. grid = (R/16, O/64), block = 32.
// ---------------------------------------------------------------------------
__global__ __launch_bounds__(32) void gemm_wT_kernel(
    const float* __restrict__ A, int Hin,
    const bf16* __restrict__ W, int ldw, int coloff,
    const float* __restrict__ bias,
    float* __restrict__ out, int O, float alpha) {
  int lane = threadIdx.x & 31;
  int half = lane >> 4, idx = lane & 15;
  int m0 = blockIdx.x * 16;
  int o0 = blockIdx.y * 64;
  v8f acc[4] = {};
  const float* arow = A + (size_t)(m0 + idx) * Hin;
  for (int kk = 0; kk < Hin; kk += 32) {
    v16bf af = load_a_f32(arow, kk, half, alpha);
#pragma unroll
    for (int t = 0; t < 4; ++t) {
      int col = o0 + t * 16 + idx;
      v16bf wf = load_frag_bf16(W + (size_t)col * ldw + coloff, kk, half);
      acc[t] = wmma_bf16(af, wf, acc[t]);
    }
  }
#pragma unroll
  for (int t = 0; t < 4; ++t) {
    int col = o0 + t * 16 + idx;
    float bb = bias ? bias[col] : 0.0f;
#pragma unroll
    for (int v = 0; v < 8; ++v) {
      int m = m0 + v + 8 * half;
      out[(size_t)m * O + col] = acc[t][v] + bb;
    }
  }
}

// ---------------------------------------------------------------------------
// Fused edge-message kernel: both AP->UE and UE->AP MLPs in one pass over e.
// Block = 128 threads (4 waves), one 16-edge-row tile (same b,l; k=k0..k0+15).
// ---------------------------------------------------------------------------
__global__ __launch_bounds__(128) void edge_msg_kernel(
    const float* __restrict__ e,
    const bf16* __restrict__ Wa1,   // a2u_w1 bf16 (256x512); use cols [256,512)
    const bf16* __restrict__ Wa2,   // a2u_w2 bf16 (256x256)
    const float* __restrict__ bA2,  // a2u_b2
    const bf16* __restrict__ Wu1,   // u2a_w1 bf16 (256x512); cols [256,512)
    const bf16* __restrict__ Wu2,   // u2a_w2
    const float* __restrict__ bU2,  // u2a_b2
    const float* __restrict__ P_ap, // (B*L,256) = h_ap@a2u_w1[:,:H]^T + a2u_b1
    const float* __restrict__ P_ue, // (B*K,256) = h_ue@u2a_w1[:,:H]^T + u2a_b1
    float* __restrict__ m_ue, float* __restrict__ m_ap) {
  __shared__ bf16 tA[16 * Hdim];
  __shared__ bf16 tU[16 * Hdim];
  int base = blockIdx.x * 16;
  int b = base / (Ldim * Kdim);
  int rem = base % (Ldim * Kdim);
  int l = rem / Kdim, k0 = rem % Kdim;
  int wv = threadIdx.x >> 5;
  int lane = threadIdx.x & 31;
  int half = lane >> 4, idx = lane & 15;
  int o0 = wv * 64;

  // ---- phase 1: pre = e @ W1_e^T  (both paths share the e fragment) ----
  v8f accA[4] = {}, accU[4] = {};
  const float* erow = e + ((size_t)base + idx) * Hdim;
  for (int kk = 0; kk < Hdim; kk += 32) {
    v16bf af = load_a_f32(erow, kk, half, 1.0f);
#pragma unroll
    for (int t = 0; t < 4; ++t) {
      int col = o0 + t * 16 + idx;
      accA[t] = wmma_bf16(af, load_frag_bf16(Wa1 + (size_t)col * 512 + 256, kk, half), accA[t]);
      accU[t] = wmma_bf16(af, load_frag_bf16(Wu1 + (size_t)col * 512 + 256, kk, half), accU[t]);
    }
  }
  const float* pap = P_ap + (size_t)(b * Ldim + l) * Hdim;
#pragma unroll
  for (int t = 0; t < 4; ++t) {
    int col = o0 + t * 16 + idx;
#pragma unroll
    for (int v = 0; v < 8; ++v) {
      int m = v + 8 * half;
      float xa = accA[t][v] + pap[col];
      tA[m * Hdim + col] = (bf16)fmaxf(xa, 0.0f);
      float xu = accU[t][v] + P_ue[(size_t)(b * Kdim + k0 + m) * Hdim + col];
      tU[m * Hdim + col] = (bf16)fmaxf(xu, 0.0f);
    }
  }
  __syncthreads();

  // ---- phase 2: msg = relu(pre) @ W2^T, then reduce ----
  v8f msgA[4] = {}, msgU[4] = {};
  const bf16* ra = tA + idx * Hdim;
  const bf16* ru = tU + idx * Hdim;
  for (int kk = 0; kk < Hdim; kk += 32) {
    v16bf aA = load_frag_bf16(ra, kk, half);
    v16bf aU = load_frag_bf16(ru, kk, half);
#pragma unroll
    for (int t = 0; t < 4; ++t) {
      int col = o0 + t * 16 + idx;
      msgA[t] = wmma_bf16(aA, load_frag_bf16(Wa2 + (size_t)col * Hdim, kk, half), msgA[t]);
      msgU[t] = wmma_bf16(aU, load_frag_bf16(Wu2 + (size_t)col * Hdim, kk, half), msgU[t]);
    }
  }
#pragma unroll
  for (int t = 0; t < 4; ++t) {
    int col = o0 + t * 16 + idx;
    float ba = bA2[col], bu = bU2[col];
    float su = 0.0f;
#pragma unroll
    for (int v = 0; v < 8; ++v) {
      int m = v + 8 * half;
      atomicAdd(&m_ue[(size_t)(b * Kdim + k0 + m) * Hdim + col], msgA[t][v] + ba);
      su += msgU[t][v] + bu;  // sum over this half's 8 rows (k values)
    }
    su += __shfl_xor(su, 16, 32);  // combine both halves (all 16 rows)
    if (half == 0)
      atomicAdd(&m_ap[(size_t)(b * Ldim + l) * Hdim + col], su);
  }
}

// ---------------------------------------------------------------------------
// Fused edge-update kernel: e_new = relu(e@ed_w1[:,2H:]^T + Qu + Qa) @ ed_w2^T + b2
// ---------------------------------------------------------------------------
__global__ __launch_bounds__(128) void edge_update_kernel(
    const float* __restrict__ e,
    const bf16* __restrict__ W1,    // ed_w1 bf16 (256x768); cols [512,768)
    const bf16* __restrict__ W2,    // ed_w2 bf16 (256x256)
    const float* __restrict__ b2,   // ed_b2
    const float* __restrict__ Q_ue, // (B*K,256) = h_ue_new@ed_w1[:,:H]^T
    const float* __restrict__ Q_ap, // (B*L,256) = h_ap_new@ed_w1[:,H:2H]^T + ed_b1
    float* __restrict__ e_new) {
  __shared__ bf16 tP[16 * Hdim];
  int base = blockIdx.x * 16;
  int b = base / (Ldim * Kdim);
  int rem = base % (Ldim * Kdim);
  int l = rem / Kdim, k0 = rem % Kdim;
  int wv = threadIdx.x >> 5;
  int lane = threadIdx.x & 31;
  int half = lane >> 4, idx = lane & 15;
  int o0 = wv * 64;

  v8f acc[4] = {};
  const float* erow = e + ((size_t)base + idx) * Hdim;
  for (int kk = 0; kk < Hdim; kk += 32) {
    v16bf af = load_a_f32(erow, kk, half, 1.0f);
#pragma unroll
    for (int t = 0; t < 4; ++t) {
      int col = o0 + t * 16 + idx;
      acc[t] = wmma_bf16(af, load_frag_bf16(W1 + (size_t)col * 768 + 512, kk, half), acc[t]);
    }
  }
  const float* qa = Q_ap + (size_t)(b * Ldim + l) * Hdim;
#pragma unroll
  for (int t = 0; t < 4; ++t) {
    int col = o0 + t * 16 + idx;
#pragma unroll
    for (int v = 0; v < 8; ++v) {
      int m = v + 8 * half;
      float x = acc[t][v] + qa[col] + Q_ue[(size_t)(b * Kdim + k0 + m) * Hdim + col];
      tP[m * Hdim + col] = (bf16)fmaxf(x, 0.0f);
    }
  }
  __syncthreads();

  v8f out[4] = {};
  const bf16* rp = tP + idx * Hdim;
  for (int kk = 0; kk < Hdim; kk += 32) {
    v16bf aP = load_frag_bf16(rp, kk, half);
#pragma unroll
    for (int t = 0; t < 4; ++t) {
      int col = o0 + t * 16 + idx;
      out[t] = wmma_bf16(aP, load_frag_bf16(W2 + (size_t)col * Hdim, kk, half), out[t]);
    }
  }
#pragma unroll
  for (int t = 0; t < 4; ++t) {
    int col = o0 + t * 16 + idx;
    float bb = b2[col];
#pragma unroll
    for (int v = 0; v < 8; ++v) {
      int m = v + 8 * half;
      e_new[(size_t)(base + m) * Hdim + col] = out[t][v] + bb;
    }
  }
}

// ---------------------------------------------------------------------------
// GRU elementwise gates: h' = (1-z)*n + z*h  (gi/gh are (R,768) precomputed)
// ---------------------------------------------------------------------------
__global__ void gru_ew_kernel(const float* __restrict__ gi,
                              const float* __restrict__ gh,
                              const float* __restrict__ h,
                              float* __restrict__ hn, int total) {
  int i = blockIdx.x * blockDim.x + threadIdx.x;
  if (i >= total) return;
  int r = i / Hdim, c = i % Hdim;
  const float* gir = gi + (size_t)r * (3 * Hdim);
  const float* ghr = gh + (size_t)r * (3 * Hdim);
  float ir = gir[c], iz = gir[Hdim + c], inn = gir[2 * Hdim + c];
  float hr = ghr[c], hz = ghr[Hdim + c], hnn = ghr[2 * Hdim + c];
  float rg = 1.0f / (1.0f + __expf(-(ir + hr)));
  float zg = 1.0f / (1.0f + __expf(-(iz + hz)));
  float ng = tanhf(inn + rg * hnn);
  hn[i] = (1.0f - zg) * ng + zg * h[i];
}

// ---------------------------------------------------------------------------
// Host launcher
// ---------------------------------------------------------------------------
extern "C" void kernel_launch(void* const* d_in, const int* in_sizes, int n_in,
                              void* d_out, int out_size, void* d_ws, size_t ws_size,
                              hipStream_t stream) {
  (void)in_sizes; (void)n_in; (void)out_size; (void)ws_size;
  const float* h_ue   = (const float*)d_in[0];
  const float* h_ap   = (const float*)d_in[1];
  const float* e      = (const float*)d_in[2];
  const float* a2u_w1 = (const float*)d_in[3];
  const float* a2u_b1 = (const float*)d_in[4];
  const float* a2u_w2 = (const float*)d_in[5];
  const float* a2u_b2 = (const float*)d_in[6];
  const float* u2a_w1 = (const float*)d_in[7];
  const float* u2a_b1 = (const float*)d_in[8];
  const float* u2a_w2 = (const float*)d_in[9];
  const float* u2a_b2 = (const float*)d_in[10];
  const float* ue_wih = (const float*)d_in[11];
  const float* ue_bih = (const float*)d_in[12];
  const float* ue_whh = (const float*)d_in[13];
  const float* ue_bhh = (const float*)d_in[14];
  const float* ap_wih = (const float*)d_in[15];
  const float* ap_bih = (const float*)d_in[16];
  const float* ap_whh = (const float*)d_in[17];
  const float* ap_bhh = (const float*)d_in[18];
  const float* ed_w1  = (const float*)d_in[19];
  const float* ed_b1  = (const float*)d_in[20];
  const float* ed_w2  = (const float*)d_in[21];
  const float* ed_b2  = (const float*)d_in[22];

  const int NK = Bdim * Kdim;          // 2048  UE rows
  const int NL = Bdim * Ldim;          // 4096  AP rows
  const int NE = Bdim * Ldim * Kdim;   // 131072 edge rows

  float* out_h_ue = (float*)d_out;                 // (B,K,H)
  float* out_h_ap = out_h_ue + (size_t)NK * Hdim;  // (B,L,H)
  float* out_e    = out_h_ue + (size_t)(NK + NL) * Hdim;  // (B,L*K,H)

  // workspace layout (floats first, bf16 weights after)
  float* wsf   = (float*)d_ws;
  float* m_ue  = wsf;                  // NK*H
  float* m_ap  = wsf + 524288;         // NL*H
  float* P_ap  = wsf + 1572864;        // NL*H
  float* P_ue  = wsf + 2621440;        // NK*H
  float* Q_ue  = wsf + 3145728;        // NK*H
  float* Q_ap  = wsf + 3670016;        // NL*H
  float* gi    = wsf + 4718592;        // NL*3H (max)
  float* gh    = wsf + 7864320;        // NL*3H (max)
  bf16* bfb     = (bf16*)(wsf + 11010048);
  bf16* a2u_w1b = bfb;                 // 131072
  bf16* u2a_w1b = bfb + 131072;        // 131072
  bf16* a2u_w2b = bfb + 262144;        // 65536
  bf16* u2a_w2b = bfb + 327680;        // 65536
  bf16* ed_w1b  = bfb + 393216;        // 196608
  bf16* ed_w2b  = bfb + 589824;        // 65536
  bf16* ue_wihb = bfb + 655360;        // 196608
  bf16* ue_whhb = bfb + 851968;        // 196608
  bf16* ap_wihb = bfb + 1048576;       // 196608
  bf16* ap_whhb = bfb + 1245184;       // 196608

  // 0) zero the atomic accumulators (m_ue + m_ap contiguous)
  {
    int n = 1572864;
    zero_kernel<<<(n + 255) / 256, 256, 0, stream>>>(m_ue, n);
  }
  // 1) convert weights to bf16
  auto cvt = [&](const float* s, bf16* d, int n) {
    cvt_bf16_kernel<<<(n + 255) / 256, 256, 0, stream>>>(s, d, n);
  };
  cvt(a2u_w1, a2u_w1b, 131072);
  cvt(u2a_w1, u2a_w1b, 131072);
  cvt(a2u_w2, a2u_w2b, 65536);
  cvt(u2a_w2, u2a_w2b, 65536);
  cvt(ed_w1,  ed_w1b, 196608);
  cvt(ed_w2,  ed_w2b, 65536);
  cvt(ue_wih, ue_wihb, 196608);
  cvt(ue_whh, ue_whhb, 196608);
  cvt(ap_wih, ap_wihb, 196608);
  cvt(ap_whh, ap_whhb, 196608);

  // 2) node projections for message MLP first linears (+b1 folded in)
  gemm_wT_kernel<<<dim3(NL / 16, Hdim / 64), 32, 0, stream>>>(
      h_ap, Hdim, a2u_w1b, 512, 0, a2u_b1, P_ap, Hdim, 1.0f);
  gemm_wT_kernel<<<dim3(NK / 16, Hdim / 64), 32, 0, stream>>>(
      h_ue, Hdim, u2a_w1b, 512, 0, u2a_b1, P_ue, Hdim, 1.0f);

  // 3) fused edge-message pass (both directions), atomic-reduced means
  edge_msg_kernel<<<NE / 16, 128, 0, stream>>>(
      e, a2u_w1b, a2u_w2b, a2u_b2, u2a_w1b, u2a_w2b, u2a_b2,
      P_ap, P_ue, m_ue, m_ap);

  // 4) GRU for UE nodes (mean scale 1/L folded into gi GEMM)
  gemm_wT_kernel<<<dim3(NK / 16, (3 * Hdim) / 64), 32, 0, stream>>>(
      m_ue, Hdim, ue_wihb, Hdim, 0, ue_bih, gi, 3 * Hdim, 1.0f / (float)Ldim);
  gemm_wT_kernel<<<dim3(NK / 16, (3 * Hdim) / 64), 32, 0, stream>>>(
      h_ue, Hdim, ue_whhb, Hdim, 0, ue_bhh, gh, 3 * Hdim, 1.0f);
  {
    int n = NK * Hdim;
    gru_ew_kernel<<<(n + 255) / 256, 256, 0, stream>>>(gi, gh, h_ue, out_h_ue, n);
  }

  // 5) GRU for AP nodes (mean scale 1/K)
  gemm_wT_kernel<<<dim3(NL / 16, (3 * Hdim) / 64), 32, 0, stream>>>(
      m_ap, Hdim, ap_wihb, Hdim, 0, ap_bih, gi, 3 * Hdim, 1.0f / (float)Kdim);
  gemm_wT_kernel<<<dim3(NL / 16, (3 * Hdim) / 64), 32, 0, stream>>>(
      h_ap, Hdim, ap_whhb, Hdim, 0, ap_bhh, gh, 3 * Hdim, 1.0f);
  {
    int n = NL * Hdim;
    gru_ew_kernel<<<(n + 255) / 256, 256, 0, stream>>>(gi, gh, h_ap, out_h_ap, n);
  }

  // 6) edge-update node projections from NEW node states
  gemm_wT_kernel<<<dim3(NK / 16, Hdim / 64), 32, 0, stream>>>(
      out_h_ue, Hdim, ed_w1b, 768, 0, nullptr, Q_ue, Hdim, 1.0f);
  gemm_wT_kernel<<<dim3(NL / 16, Hdim / 64), 32, 0, stream>>>(
      out_h_ap, Hdim, ed_w1b, 768, 256, ed_b1, Q_ap, Hdim, 1.0f);

  // 7) fused edge update -> e_new
  edge_update_kernel<<<NE / 16, 128, 0, stream>>>(
      e, ed_w1b, ed_w2b, ed_b2, Q_ue, Q_ap, out_e);
}
